// MyLSTM_20925080666767
// MI455X (gfx1250) — matrix operand order repaired and checked by reading, not compiled, not run on hardware
//
#include <hip/hip_runtime.h>
#include <hip/hip_bf16.h>

// ---------------------------------------------------------------------------
// LSTM (2 layers, B=32, T=512, I=256, H=1024) + FC (O=512, T truncated to 500)
// MI455X / gfx1250:
//  - bf16 WMMA (v_wmma_f32_16x16x32_bf16) for all GEMMs, f32 accumulate
//  - batched GEMMs: 128x128 block tiles staged via LDS, 4x2 register blocking
//    per wave (~64 FLOP/B of L2 traffic instead of ~8)
//  - recurrent scan: persistent kernel, W_hh slice (128KB) resident in LDS
//    (possible only with CDNA5's 320KB/WGP), device-scope step barrier via
//    agent-scope atomics, global_prefetch for next step's xg tile
// ---------------------------------------------------------------------------

typedef __attribute__((ext_vector_type(16))) __bf16 v16bf;
typedef __attribute__((ext_vector_type(8)))  float  v8f;

union Frag { uint4 u[2]; v16bf v; };

__device__ __forceinline__ unsigned short f32_to_bf16(float f) {
    union { float f; unsigned u; } x; x.f = f;
    unsigned r = x.u + 0x7FFFu + ((x.u >> 16) & 1u);   // round-to-nearest-even
    return (unsigned short)(r >> 16);
}

__device__ __forceinline__ float fast_sigmoid(float x) {
    return 1.0f / (1.0f + __expf(-x));
}
__device__ __forceinline__ float fast_tanh(float x) {
    float e = __expf(2.0f * x);
    return (e - 1.0f) / (e + 1.0f);
}

// --------------------------- small utility kernels -------------------------

__global__ __launch_bounds__(256) void f2bf_kernel(const float* __restrict__ s,
                                                   unsigned short* __restrict__ d, int n) {
    int i = blockIdx.x * 256 + threadIdx.x;
    if (i < n) d[i] = f32_to_bf16(s[i]);
}

__global__ __launch_bounds__(256) void bias_comb_kernel(const float* __restrict__ a,
                                                        const float* __restrict__ b,
                                                        float* __restrict__ d, int n) {
    int i = blockIdx.x * 256 + threadIdx.x;
    if (i < n) d[i] = a[i] + b[i];
}

__global__ void zero_kernel(unsigned* __restrict__ p, int n) {
    int i = blockIdx.x * blockDim.x + threadIdx.x;
    if (i < n) p[i] = 0u;
}

// --------------------------- tiled WMMA GEMM -------------------------------
// C[M][ldC] = A * W^T + bias ; A bf16 (layout by MODE), W [N][K] bf16
// row-major, bias f32[N], C f32.
// Block = 256 threads = 8 waves -> 128x128 C tile. Per K-chunk of 32:
// stage A(128x32) + B(128x32) bf16 in LDS (8KB+8KB), each wave computes a
// 4(M) x 2(N) grid of 16x16 WMMA tiles from LDS fragments.
// MODE 0: A = x  [32][512][K]   row = t*32 + b
// MODE 1: A row-major [M][K]
// MODE 2: A = h2 [512][32][K]   row = b*500 + t   (FC output mapping)
template<int MODE>
__global__ __launch_bounds__(256) void wmma_gemm_kernel(
    const unsigned short* __restrict__ A,
    const unsigned short* __restrict__ W,
    const float* __restrict__ bias,
    float* __restrict__ C,
    int nBlocksN, int K, int ldC)
{
    __shared__ unsigned short ldsA[128 * 32];   // 8 KB
    __shared__ unsigned short ldsB[128 * 32];   // 8 KB

    int bm   = blockIdx.x / nBlocksN;
    int bn   = blockIdx.x % nBlocksN;
    int tid  = threadIdx.x;
    int lane = tid & 31;
    int wave = tid >> 5;
    int wm = wave & 1;          // M half   (64 rows)
    int wn = wave >> 1;         // N quarter (32 cols)
    int nl = lane & 15;
    int koffA = (lane < 16) ? 0 : 8;
    int koffB = (lane < 16) ? 0 : 8;

    v8f acc[4][2];
#pragma unroll
    for (int i = 0; i < 4; ++i)
#pragma unroll
        for (int j = 0; j < 2; ++j) acc[i][j] = (v8f){};

    int ktiles = K >> 5;
    for (int kt = 0; kt < ktiles; ++kt) {
        int kb = kt << 5;
        // ---- stage 128x32 A and B chunks into LDS (2 uint4 each per thread)
#pragma unroll
        for (int s = 0; s < 2; ++s) {
            int id = tid + s * 256;
            int r  = id >> 2;
            int c  = (id & 3) << 3;
            int row = bm * 128 + r;
            size_t a_base;
            if (MODE == 0) {            // x[B=32][T=512][K]
                int b = row & 31, t = row >> 5;
                a_base = ((size_t)b * 512 + t) * (size_t)K;
            } else if (MODE == 1) {     // row-major
                a_base = (size_t)row * (size_t)K;
            } else {                    // h2[T=512][B=32][K], row = b*500+t
                int b = row / 500, t = row % 500;
                a_base = ((size_t)t * 32 + b) * (size_t)K;
            }
            *(uint4*)(ldsA + r * 32 + c) = *(const uint4*)(A + a_base + kb + c);
            int ncol = bn * 128 + r;
            *(uint4*)(ldsB + r * 32 + c) =
                *(const uint4*)(W + (size_t)ncol * (size_t)K + kb + c);
        }
        __syncthreads();

        // ---- fragments from LDS, 8 WMMAs
        Frag af[4], bf[2];
#pragma unroll
        for (int i = 0; i < 4; ++i) {
            int ar = wm * 64 + i * 16 + nl;
            af[i].u[0] = *(const uint4*)(ldsA + ar * 32 + koffA);
            af[i].u[1] = *(const uint4*)(ldsA + ar * 32 + koffA + 16);
        }
#pragma unroll
        for (int j = 0; j < 2; ++j) {
            int br = wn * 32 + j * 16 + nl;
            bf[j].u[0] = *(const uint4*)(ldsB + br * 32 + koffB);
            bf[j].u[1] = *(const uint4*)(ldsB + br * 32 + koffB + 16);
        }
#pragma unroll
        for (int i = 0; i < 4; ++i)
#pragma unroll
            for (int j = 0; j < 2; ++j)
                acc[i][j] = __builtin_amdgcn_wmma_f32_16x16x32_bf16(
                    false, af[i].v, false, bf[j].v, (short)0, acc[i][j],
                    false, false);
        __syncthreads();
    }

    // ---- epilogue: bias + store
    int rbase = (lane >> 4) << 3;       // lanes 16-31 hold rows M+8
#pragma unroll
    for (int j = 0; j < 2; ++j) {
        int ncol = bn * 128 + wn * 32 + j * 16 + nl;
        float bv = bias[ncol];
#pragma unroll
        for (int i = 0; i < 4; ++i) {
#pragma unroll
            for (int r = 0; r < 8; ++r) {
                size_t crow = (size_t)(bm * 128 + wm * 64 + i * 16 + rbase + r);
                C[crow * (size_t)ldC + ncol] = acc[i][j][r] + bv;
            }
        }
    }
}

// --------------------------- persistent LSTM scan --------------------------
// 64 WGs x 256 threads. WG w owns H columns [16w,16w+16). LDS holds the
// W_hh rows for all 4 gates at that slice (64 rows x 1024 K bf16 = 128KB).
// Wave v (0..7): gate g = v&3, m-tile m = v>>2 (batch rows 16m..16m+15).
// Per step: gates = xg[t] + h_{t-1} @ Whh_slice^T (WMMA), then 2 update
// waves apply the LSTM cell and write h_t (bf16) for the next step / layer.
#define SCAN_WGS 64

__global__ __launch_bounds__(256) void lstm_scan_kernel(
    const float* __restrict__ xg,            // [512][32][4096] f32
    const unsigned short* __restrict__ Whh,  // [4096][1024] bf16
    unsigned short* __restrict__ h_seq,      // [512][32][1024] bf16 out
    unsigned* __restrict__ ctr)
{
    __shared__ unsigned short wlds[64 * 1024];     // 128 KB
    __shared__ float gates[8][16][16];             // 8 KB
    __shared__ float c_state[2][16][16];           // 2 KB

    int w    = blockIdx.x;
    int tid  = threadIdx.x;
    int lane = tid & 31;
    int wave = tid >> 5;
    int g = wave & 3, m = wave >> 2;
    int nl = lane & 15;

    // preload W_hh slice: local row r = g2*16 + c  <-  Whh[g2*1024 + 16w + c]
    for (int r4 = tid; r4 < 64 * 128; r4 += 256) {   // 128 uint4-chunks per row
        int r  = r4 >> 7;
        int kc = (r4 & 127) << 3;
        int g2 = r >> 4, c = r & 15;
        const uint4* src =
            (const uint4*)(Whh + ((size_t)(g2 * 1024 + w * 16 + c)) * 1024 + kc);
        *(uint4*)(wlds + r * 1024 + kc) = *src;
    }
    for (int i = tid; i < 512; i += 256) ((float*)c_state)[i] = 0.0f;
    __syncthreads();

    int mrow_base = m * 16 + ((lane >> 4) << 3);
    int ncol      = g * 1024 + w * 16 + nl;
    int arow      = m * 16 + nl;
    int koffA     = (lane < 16) ? 0 : 8;
    int koffB     = (lane < 16) ? 0 : 8;
    const unsigned short* wrow = wlds + (g * 16 + nl) * 1024;

    for (int t = 0; t < 512; ++t) {
        // acc <- xg tile (input contribution, biases already folded in)
        v8f acc;
        const float* xgp = xg + ((size_t)t * 32 + mrow_base) * 4096 + ncol;
#pragma unroll
        for (int r = 0; r < 8; ++r) acc[r] = xgp[(size_t)r * 4096];
        if (t < 511)   // hide next step's xg read behind the step barrier
            __builtin_prefetch(xgp + (size_t)32 * 4096, 0, 1);

        if (t > 0) {
            const unsigned short* hprev =
                h_seq + (size_t)(t - 1) * 32 * 1024 + (size_t)arow * 1024;
            for (int kt = 0; kt < 32; ++kt) {
                int kb = kt << 5;
                Frag a, b;
                a.u[0] = *(const uint4*)(hprev + kb + koffA);
                a.u[1] = *(const uint4*)(hprev + kb + koffA + 16);
                b.u[0] = *(const uint4*)(wrow + kb + koffB);   // LDS b128
                b.u[1] = *(const uint4*)(wrow + kb + koffB + 16);
                acc = __builtin_amdgcn_wmma_f32_16x16x32_bf16(
                          false, a.v, false, b.v, (short)0, acc, false, false);
            }
        }

        // stage gate tile to LDS
        int rb = (lane >> 4) << 3;
#pragma unroll
        for (int r = 0; r < 8; ++r) gates[wave][rb + r][nl] = acc[r];
        __syncthreads();

        // waves 0,1 apply LSTM cell for m-tile = wave
        if (wave < 2) {
            int mm  = wave;
            int col = lane & 15;
#pragma unroll
            for (int j = 0; j < 8; ++j) {
                int r = rb + j;
                float gi = gates[mm * 4 + 0][r][col];
                float gf = gates[mm * 4 + 1][r][col];
                float gg = gates[mm * 4 + 2][r][col];
                float go = gates[mm * 4 + 3][r][col];
                float c  = fast_sigmoid(gf) * c_state[mm][r][col]
                         + fast_sigmoid(gi) * fast_tanh(gg);
                c_state[mm][r][col] = c;
                float h = fast_sigmoid(go) * fast_tanh(c);
                h_seq[((size_t)t * 32 + mm * 16 + r) * 1024 + w * 16 + col] =
                    f32_to_bf16(h);
            }
        }
        __syncthreads();

        // device-wide step barrier: release h_t, wait for all 64 WGs
        if (tid == 0) {
            __hip_atomic_fetch_add(ctr, 1u, __ATOMIC_RELEASE,
                                   __HIP_MEMORY_SCOPE_AGENT);
            unsigned target = (unsigned)(t + 1) * (unsigned)gridDim.x;
            while (__hip_atomic_load(ctr, __ATOMIC_ACQUIRE,
                                     __HIP_MEMORY_SCOPE_AGENT) < target)
                __builtin_amdgcn_s_sleep(2);
        }
        __syncthreads();
    }
}

// ------------------------------- launcher ----------------------------------

extern "C" void kernel_launch(void* const* d_in, const int* in_sizes, int n_in,
                              void* d_out, int out_size, void* d_ws, size_t ws_size,
                              hipStream_t stream) {
    const float* x     = (const float*)d_in[0];
    const float* W_ih0 = (const float*)d_in[1];
    const float* W_hh0 = (const float*)d_in[2];
    const float* b_ih0 = (const float*)d_in[3];
    const float* b_hh0 = (const float*)d_in[4];
    const float* W_ih1 = (const float*)d_in[5];
    const float* W_hh1 = (const float*)d_in[6];
    const float* b_ih1 = (const float*)d_in[7];
    const float* b_hh1 = (const float*)d_in[8];
    const float* W_fc  = (const float*)d_in[9];
    const float* b_fc  = (const float*)d_in[10];
    float* out = (float*)d_out;

    char* ws = (char*)d_ws;
    size_t off = 0;
    auto carve = [&](size_t bytes) -> void* {
        void* p = ws + off;
        off += (bytes + 255) & ~(size_t)255;
        return p;
    };
    unsigned short* xbf   = (unsigned short*)carve((size_t)32 * 512 * 256 * 2);
    unsigned short* wih0  = (unsigned short*)carve((size_t)4096 * 256 * 2);
    unsigned short* whh0  = (unsigned short*)carve((size_t)4096 * 1024 * 2);
    unsigned short* wih1  = (unsigned short*)carve((size_t)4096 * 1024 * 2);
    unsigned short* whh1  = (unsigned short*)carve((size_t)4096 * 1024 * 2);
    unsigned short* wfc   = (unsigned short*)carve((size_t)512 * 1024 * 2);
    float*          bias0 = (float*)carve(4096 * 4);
    float*          bias1 = (float*)carve(4096 * 4);
    unsigned*       ctr   = (unsigned*)carve(256);
    unsigned short* h1    = (unsigned short*)carve((size_t)512 * 32 * 1024 * 2);
    unsigned short* h2    = (unsigned short*)carve((size_t)512 * 32 * 1024 * 2);
    float*          xg    = (float*)carve((size_t)512 * 32 * 4096 * 4);

    auto cvt = [&](const float* s, unsigned short* d, int n) {
        f2bf_kernel<<<(n + 255) / 256, 256, 0, stream>>>(s, d, n);
    };
    cvt(x,     xbf,  32 * 512 * 256);
    cvt(W_ih0, wih0, 4096 * 256);
    cvt(W_hh0, whh0, 4096 * 1024);
    cvt(W_ih1, wih1, 4096 * 1024);
    cvt(W_hh1, whh1, 4096 * 1024);
    cvt(W_fc,  wfc,  512 * 1024);
    bias_comb_kernel<<<16, 256, 0, stream>>>(b_ih0, b_hh0, bias0, 4096);
    bias_comb_kernel<<<16, 256, 0, stream>>>(b_ih1, b_hh1, bias1, 4096);
    zero_kernel<<<1, 64, 0, stream>>>(ctr, 64);

    // Layer 0: xg = x @ W_ih0^T + bias0  (M=16384, N=4096, K=256)
    wmma_gemm_kernel<0><<<128 * 32, 256, 0, stream>>>(
        xbf, wih0, bias0, xg, 32, 256, 4096);
    lstm_scan_kernel<<<SCAN_WGS, 256, 0, stream>>>(xg, whh0, h1, ctr);

    // Layer 1: xg = h1 @ W_ih1^T + bias1 (M=16384, N=4096, K=1024)
    wmma_gemm_kernel<1><<<128 * 32, 256, 0, stream>>>(
        h1, wih1, bias1, xg, 32, 1024, 4096);
    lstm_scan_kernel<<<SCAN_WGS, 256, 0, stream>>>(xg, whh1, h2, ctr + 32);

    // FC: out[b][t][o] = h2[t][b][:] @ W_fc^T + b_fc (M=16000, N=512, K=1024)
    wmma_gemm_kernel<2><<<125 * 4, 256, 0, stream>>>(
        h2, wfc, b_fc, out, 4, 1024, 512);
}
